// Attention_64493228917004
// MI455X (gfx1250) — matrix operand order repaired and checked by reading, not compiled, not run on hardware
//
#include <hip/hip_runtime.h>

#define D_MODEL 1024
#define SEQ     4096
#define HEADS   16
#define DK      64

typedef __attribute__((ext_vector_type(16))) __bf16 v16bf;
typedef __attribute__((ext_vector_type(8)))  __bf16 v8bf;
typedef __attribute__((ext_vector_type(8)))  float  v8f;
typedef unsigned short ushort_t;

// ---------- helpers ----------

__device__ __forceinline__ ushort_t f2bf(float f) {
    unsigned u = __float_as_uint(f);
    u += 0x7FFFu + ((u >> 16) & 1u);           // round-to-nearest-even
    return (ushort_t)(u >> 16);
}

// A-fragment (16x32 bf16, MxK) from row-major bf16 [.., ld]:
// lane<16: elems 0..15 = K k0..k0+15
// lane>=16: +16 on the K base (contiguous, matches B fragments)
__device__ __forceinline__ v16bf load_a16(const ushort_t* __restrict__ base,
                                          int ld, int row, int k0, int lane) {
    const ushort_t* p = base + row * ld + k0 + ((lane >> 4) << 4);
    v8bf lo = *(const v8bf*)p;
    v8bf hi = *(const v8bf*)(p + 8);
    v16bf a;
#pragma unroll
    for (int i = 0; i < 8; ++i) { a[i] = lo[i]; a[i + 8] = hi[i]; }
    return a;
}

// B-fragment (32x16 bf16, KxN) where B's column n lives as a contiguous
// 16-element run in memory: lane holds column (lane&15), elements are
// K = (lane>=16 ? 16 : 0) + e  (contiguous, same k order as load_a16).
__device__ __forceinline__ v16bf load_bcol(const ushort_t* __restrict__ base,
                                           int ld, int rowbase, int colbase, int lane) {
    return *(const v16bf*)(base + (rowbase + (lane & 15)) * ld
                                + colbase + ((lane >> 4) << 4));
}

// Pre-packed weight B-fragment: one aligned 32B load per lane.
__device__ __forceinline__ v16bf load_b_pack(const ushort_t* __restrict__ wp,
                                             int kt, int nt, int lane) {
    const int NT = D_MODEL / 16;
    return *(const v16bf*)(wp + ((((size_t)kt * NT + nt) * 32 + lane) << 4));
}

__device__ __forceinline__ v8f wmma_bf16(v16bf a, v16bf b, v8f c) {
    return __builtin_amdgcn_wmma_f32_16x16x32_bf16(false, a, false, b,
                                                   (short)0, c, false, false);
}

// ---------- kernel 1: x fp32 -> bf16 ----------

__global__ __launch_bounds__(256) void conv_x(const float* __restrict__ x,
                                              ushort_t* __restrict__ xb) {
    int i = (blockIdx.x * 256 + threadIdx.x) * 4;
    float4 v = *(const float4*)(x + i);
    unsigned lo = (unsigned)f2bf(v.x) | ((unsigned)f2bf(v.y) << 16);
    unsigned hi = (unsigned)f2bf(v.z) | ((unsigned)f2bf(v.w) << 16);
    uint2 u; u.x = lo; u.y = hi;
    *(uint2*)(xb + i) = u;
}

// ---------- kernel 2: pack W (torch Linear: y = x @ W^T, contract over d) ----
// Packed layout: tile (kt, nt) -> 32 lanes x 16 bf16 each, lane holds
// output-column o = nt*16 + (lane&15), K run d = kt*32 + (lane>=16?16:0) + e.

__global__ __launch_bounds__(256) void pack_w(const float* __restrict__ w0, const float* __restrict__ w1,
                                              const float* __restrict__ w2, const float* __restrict__ w3,
                                              ushort_t* __restrict__ p0, ushort_t* __restrict__ p1,
                                              ushort_t* __restrict__ p2, ushort_t* __restrict__ p3) {
    const float* w; ushort_t* p;
    switch (blockIdx.y) {
        case 0:  w = w0; p = p0; break;
        case 1:  w = w1; p = p1; break;
        case 2:  w = w2; p = p2; break;
        default: w = w3; p = p3; break;
    }
    int tid  = blockIdx.x * 256 + threadIdx.x;   // 65536 threads
    int lane = tid & 31;
    int tile = tid >> 5;                          // (D/32)*(D/16) = 2048 tiles
    const int NT = D_MODEL / 16;
    int nt = tile % NT;
    int kt = tile / NT;
    int orow  = nt * 16 + (lane & 15);
    int dbase = kt * 32 + ((lane >> 4) << 4);
    const float* src = w + (size_t)orow * D_MODEL + dbase;
    ushort_t* dst = p + (((size_t)tile * 32 + lane) << 4);
#pragma unroll
    for (int e = 0; e < 16; ++e) dst[e] = f2bf(src[e]);
}

// ---------- kernel 3: Q/K/V GEMM (one wave = 16x64 strip) ----------
// mode 0: Q (scaled by 1/sqrt(dk), row-major bf16)
// mode 1: K (row-major bf16)
// mode 2: V (stored transposed: Vt[d, s] bf16)

__global__ __launch_bounds__(256) void gemm_qkv(const ushort_t* __restrict__ xb,
                                                const ushort_t* __restrict__ wq,
                                                const ushort_t* __restrict__ wk,
                                                const ushort_t* __restrict__ wv,
                                                ushort_t* __restrict__ Qb,
                                                ushort_t* __restrict__ Kb,
                                                ushort_t* __restrict__ Vt) {
    const int lane = threadIdx.x & 31;
    const int wave = threadIdx.x >> 5;
    const int mode = blockIdx.y;
    const ushort_t* wp = (mode == 0) ? wq : (mode == 1) ? wk : wv;
    const int NT64 = D_MODEL / 64;                 // 16
    const int gw = blockIdx.x * 8 + wave;          // 4096 waves
    const int m0 = (gw / NT64) * 16;
    const int nb = (gw % NT64) * 4;                // base 16-wide n-tile
    const int row = m0 + (lane & 15);

    v8f acc[4] = {};
    for (int kt = 0; kt < D_MODEL / 32; ++kt) {
        v16bf a = load_a16(xb, D_MODEL, row, kt * 32, lane);
#pragma unroll
        for (int n = 0; n < 4; ++n) {
            v16bf b = load_b_pack(wp, kt, nb + n, lane);
            acc[n] = wmma_bf16(a, b, acc[n]);
        }
    }

    const int goff = (lane >> 4) << 3;             // +8 rows for upper half-wave
    const int ncol = lane & 15;
    if (mode == 2) {
#pragma unroll
        for (int n = 0; n < 4; ++n) {
            int d = (nb + n) * 16 + ncol;
#pragma unroll
            for (int r = 0; r < 8; ++r) {
                int m = m0 + r + goff;
                Vt[(size_t)d * SEQ + m] = f2bf(acc[n][r]);
            }
        }
    } else {
        ushort_t* out = (mode == 0) ? Qb : Kb;
        float scale = (mode == 0) ? 0.125f : 1.0f;  // 1/sqrt(64) folded into Q
#pragma unroll
        for (int n = 0; n < 4; ++n) {
            int d = (nb + n) * 16 + ncol;
#pragma unroll
            for (int r = 0; r < 8; ++r) {
                int m = m0 + r + goff;
                out[(size_t)m * D_MODEL + d] = f2bf(acc[n][r] * scale);
            }
        }
    }
}

// ---------- kernel 4: flash attention (one wave = one head x 16 rows) -------

__global__ __launch_bounds__(256) void attn(const ushort_t* __restrict__ Qb,
                                            const ushort_t* __restrict__ Kb,
                                            const ushort_t* __restrict__ Vt,
                                            ushort_t* __restrict__ Ob) {
    __shared__ ushort_t Pl[8][16 * 32];            // per-wave P transpose tile
    const int lane = threadIdx.x & 31;
    const int wave = threadIdx.x >> 5;
    const int gw = blockIdx.x * 8 + wave;          // 4096 waves
    const int h  = gw >> 8;                        // 16 heads
    const int m0 = (gw & 255) * 16;                // 256 row-blocks
    const int hb = h * DK;
    const int row = m0 + (lane & 15);
    const int prow = lane & 15;
    const int pg   = lane >> 4;

    v16bf q0 = load_a16(Qb, D_MODEL, row, hb,      lane);
    v16bf q1 = load_a16(Qb, D_MODEL, row, hb + 32, lane);

    float mr[8], lr[8];
    v8f o[4] = {};
#pragma unroll
    for (int r = 0; r < 8; ++r) { mr[r] = -1e30f; lr[r] = 0.0f; }

    ushort_t* pl = &Pl[wave][0];

    for (int t0 = 0; t0 < SEQ; t0 += 32) {
        // ---- S = Q @ K^T for a 16x32 logit tile (two 16x16 C tiles) ----
        v8f z = {};
        v16bf b00 = load_bcol(Kb, D_MODEL, t0,      hb,      lane);
        v16bf b01 = load_bcol(Kb, D_MODEL, t0,      hb + 32, lane);
        v8f s0 = wmma_bf16(q0, b00, z);
        s0     = wmma_bf16(q1, b01, s0);
        v16bf b10 = load_bcol(Kb, D_MODEL, t0 + 16, hb,      lane);
        v16bf b11 = load_bcol(Kb, D_MODEL, t0 + 16, hb + 32, lane);
        v8f s1 = wmma_bf16(q0, b10, z);
        s1     = wmma_bf16(q1, b11, s1);

        if (t0 + 32 < SEQ) {
            __builtin_prefetch(Kb + (size_t)(t0 + 32) * D_MODEL + hb, 0, 3);
            __builtin_prefetch(Vt + (size_t)hb * SEQ + (t0 + 32), 0, 3);
        }

        // ---- online softmax (rows live across 16-lane groups) ----
#pragma unroll
        for (int r = 0; r < 8; ++r) {
            float t = fmaxf(s0[r], s1[r]);
            t = fmaxf(t, __shfl_xor(t, 1));
            t = fmaxf(t, __shfl_xor(t, 2));
            t = fmaxf(t, __shfl_xor(t, 4));
            t = fmaxf(t, __shfl_xor(t, 8));
            float mn   = fmaxf(mr[r], t);
            float corr = __expf(mr[r] - mn);
            mr[r] = mn;
            float p0 = __expf(s0[r] - mn);
            float p1 = __expf(s1[r] - mn);
            s0[r] = p0; s1[r] = p1;
            float ls = p0 + p1;
            ls += __shfl_xor(ls, 1);
            ls += __shfl_xor(ls, 2);
            ls += __shfl_xor(ls, 4);
            ls += __shfl_xor(ls, 8);
            lr[r] = lr[r] * corr + ls;
#pragma unroll
            for (int n = 0; n < 4; ++n) o[n][r] = o[n][r] * corr;
        }

        // ---- transpose P (C layout -> A layout) through LDS ----
#pragma unroll
        for (int r = 0; r < 8; ++r) {
            int rw = r + (pg << 3);
            pl[rw * 32 + prow]      = f2bf(s0[r]);
            pl[rw * 32 + 16 + prow] = f2bf(s1[r]);
        }
        asm volatile("s_wait_dscnt 0" ::: "memory");
        const ushort_t* pp = pl + prow * 32 + (pg << 4);
        v8bf plo = *(const v8bf*)pp;
        v8bf phi = *(const v8bf*)(pp + 8);
        v16bf aP;
#pragma unroll
        for (int i = 0; i < 8; ++i) { aP[i] = plo[i]; aP[8 + i] = phi[i]; }

        // ---- O += P @ V  (V columns are contiguous rows of Vt) ----
#pragma unroll
        for (int n = 0; n < 4; ++n) {
            v16bf bv = load_bcol(Vt, SEQ, hb + n * 16, t0, lane);
            o[n] = wmma_bf16(aP, bv, o[n]);
        }
    }

    // ---- normalize and store O (bf16 row-major) ----
#pragma unroll
    for (int r = 0; r < 8; ++r) {
        float inv = 1.0f / lr[r];
        int m = m0 + r + (pg << 3);
#pragma unroll
        for (int n = 0; n < 4; ++n) {
            Ob[(size_t)m * D_MODEL + hb + n * 16 + prow] = f2bf(o[n][r] * inv);
        }
    }
}

// ---------- kernel 5: out = O @ Wo^T (fp32 output) ----------

__global__ __launch_bounds__(256) void gemm_out(const ushort_t* __restrict__ Ob,
                                                const ushort_t* __restrict__ wo,
                                                float* __restrict__ out) {
    const int lane = threadIdx.x & 31;
    const int wave = threadIdx.x >> 5;
    const int NT64 = D_MODEL / 64;
    const int gw = blockIdx.x * 8 + wave;
    const int m0 = (gw / NT64) * 16;
    const int nb = (gw % NT64) * 4;
    const int row = m0 + (lane & 15);

    v8f acc[4] = {};
    for (int kt = 0; kt < D_MODEL / 32; ++kt) {
        v16bf a = load_a16(Ob, D_MODEL, row, kt * 32, lane);
#pragma unroll
        for (int n = 0; n < 4; ++n) {
            v16bf b = load_b_pack(wo, kt, nb + n, lane);
            acc[n] = wmma_bf16(a, b, acc[n]);
        }
    }
    const int goff = (lane >> 4) << 3;
    const int ncol = lane & 15;
#pragma unroll
    for (int n = 0; n < 4; ++n) {
        int d = (nb + n) * 16 + ncol;
#pragma unroll
        for (int r = 0; r < 8; ++r) {
            int m = m0 + r + goff;
            out[(size_t)m * D_MODEL + d] = acc[n][r];
        }
    }
}

// ---------- host launcher ----------

extern "C" void kernel_launch(void* const* d_in, const int* in_sizes, int n_in,
                              void* d_out, int out_size, void* d_ws, size_t ws_size,
                              hipStream_t stream) {
    (void)in_sizes; (void)n_in; (void)out_size; (void)ws_size;
    const float* x  = (const float*)d_in[0];
    const float* Wq = (const float*)d_in[1];
    const float* Wk = (const float*)d_in[2];
    const float* Wv = (const float*)d_in[3];
    const float* Wo = (const float*)d_in[4];
    float* out = (float*)d_out;
    char* ws = (char*)d_ws;

    const size_t MB = 1u << 20;
    ushort_t* xb  = (ushort_t*)(ws);                 // 8 MiB (reused as Ob later)
    ushort_t* wqp = (ushort_t*)(ws +  8 * MB);       // 2 MiB
    ushort_t* wkp = (ushort_t*)(ws + 10 * MB);       // 2 MiB
    ushort_t* wvp = (ushort_t*)(ws + 12 * MB);       // 2 MiB
    ushort_t* wop = (ushort_t*)(ws + 14 * MB);       // 2 MiB
    ushort_t* Qb  = (ushort_t*)(ws + 16 * MB);       // 8 MiB
    ushort_t* Kb  = (ushort_t*)(ws + 24 * MB);       // 8 MiB
    ushort_t* Vt  = (ushort_t*)(ws + 32 * MB);       // 8 MiB
    ushort_t* Ob  = xb;                              // xb dead after gemm_qkv

    conv_x  <<<(SEQ * D_MODEL) / (256 * 4), 256, 0, stream>>>(x, xb);
    pack_w  <<<dim3(256, 4), 256, 0, stream>>>(Wq, Wk, Wv, Wo, wqp, wkp, wvp, wop);
    gemm_qkv<<<dim3(512, 3), 256, 0, stream>>>(xb, wqp, wkp, wvp, Qb, Kb, Vt);
    attn    <<<512, 256, 0, stream>>>(Qb, Kb, Vt, Ob);
    gemm_out<<<512, 256, 0, stream>>>(Ob, wop, out);
}
